// FPLSTM_Layer_18708877541856
// MI455X (gfx1250) — compile-verified, matrix operand order checked by reading, not run on hardware
//
#include <hip/hip_runtime.h>

// Problem dims (from reference): S=512, B=128, I=1024, H=1024
#define S_DIM 512
#define B_DIM 128
#define I_DIM 1024
#define H_DIM 1024
#define G5 (5 * H_DIM)   // 5120
#define G4 (4 * H_DIM)   // 4096

typedef __attribute__((ext_vector_type(16))) __bf16 bf16x16;
typedef __attribute__((ext_vector_type(8)))  __bf16 bf16x8;
typedef __attribute__((ext_vector_type(8)))  float  floatx8;
typedef __attribute__((ext_vector_type(4)))  unsigned int u32x4;
typedef __attribute__((ext_vector_type(8)))  int i32x8;
typedef __attribute__((ext_vector_type(4)))  int i32x4;

// ---- TDM-staged B tile: 64 rows x 1024 K of bf16, LDS-padded ----
// TDM pad: 4 dwords (16B) inserted after every 256 dwords (1024B) stored.
// Row = 2048B data + 2x16B pads -> 2080B stride; mid-row pad after 1024B.
#define BT_ROWS 64
#define BT_ROWSTRIDE 2080u
#define BT_BYTES (BT_ROWS * BT_ROWSTRIDE)   // 133120 B of LDS

__device__ __forceinline__ float sigmoidf_fast(float x) {
  return 1.0f / (1.0f + __expf(-x));
}

// Issue one Tensor-Data-Mover load: 64x1024 bf16 tile (row-major, row stride
// 1024 elements) from global into LDS at byte offset lds_off, with padding.
// D# packing per CDNA5 ISA.
// Toolchain: 6-arg builtin (clang-23 / therock-10.0 form).
__device__ __forceinline__ void tdm_load_weight_tile_64x1024(const __bf16* gsrc,
                                                             unsigned int lds_off) {
  unsigned long long ga = (unsigned long long)(uintptr_t)gsrc;
  u32x4 g0;
  g0[0] = 1u;                                            // count=1, user mode
  g0[1] = lds_off;                                       // lds_addr [63:32]
  g0[2] = (unsigned int)(ga & 0xffffffffull);            // global_addr[31:0]
  g0[3] = (unsigned int)((ga >> 32) & 0x01ffffffull)     // global_addr[56:32]
          | (2u << 30);                                  // type=2 (image)
  i32x8 g1;
  g1[0] = (int)((1u << 16)      // data_size=1 (2 bytes)
              | (1u << 20)      // pad_enable
              | (7u << 22)      // pad_interval: 256 dwords
              | (3u << 25));    // pad_amount: 4 dwords (16B)
  g1[1] = (int)(1024u << 16);   // tensor_dim0 = 1024 (low 16 in [31:16])
  g1[2] = (int)(64u << 16);     // tensor_dim0 hi16=0; tensor_dim1=64 (low16)
  g1[3] = (int)(1024u << 16);   // tensor_dim1 hi16=0; tile_dim0 = 1024
  g1[4] = (int)64;              // tile_dim1 = 64; tile_dim2 = 0
  g1[5] = (int)1024;            // tensor_dim0_stride = 1024 (low 32)
  g1[6] = 0;                    // stride hi16 = 0; tensor_dim1_stride = 0
  g1[7] = 0;
  i32x4 g2 = {0, 0, 0, 0};      // 2D tensor: groups 2/3 unused
  i32x4 g3 = {0, 0, 0, 0};
  i32x8 g4 = {0, 0, 0, 0, 0, 0, 0, 0};
  __builtin_amdgcn_tensor_load_to_lds(g0, g1, g2, g3, g4, 0);
}

// Load one 16x32 bf16 WMMA operand tile from a row-major GLOBAL matrix.
// ISA 16-bit A layout (wave32): lanes 0-15 hold row M=lane, K={0..7,16..23};
// lanes 16-31 hold row M=lane-16, K={8..15,24..31}. Two b128 loads per lane.
__device__ __forceinline__ bf16x16 load_tile16x32(const __bf16* __restrict__ base,
                                                  int ld, int row0, int k0, int lane) {
  const int r  = row0 + (lane & 15);
  const int kc = k0 + ((lane >> 4) << 3);
  const __bf16* p = base + (size_t)r * ld + kc;
  bf16x8 lo = *(const bf16x8*)(p);
  bf16x8 hi = *(const bf16x8*)(p + 16);
  bf16x16 out;
#pragma unroll
  for (int i = 0; i < 8; ++i) { out[i] = lo[i]; out[i + 8] = hi[i]; }
  return out;
}

// Same operand fetch but from the TDM-padded LDS tile (ds_load_b128 x2).
__device__ __forceinline__ bf16x16 lds_tile16x32(const char* sB, int row0, int k0,
                                                 int lane) {
  const int r  = row0 + (lane & 15);
  const int k1 = k0 + ((lane >> 4) << 3);
  const int k2 = k1 + 16;
  const unsigned int o1 =
      (unsigned int)r * BT_ROWSTRIDE + (unsigned int)k1 * 2u + ((k1 >= 512) ? 16u : 0u);
  const unsigned int o2 =
      (unsigned int)r * BT_ROWSTRIDE + (unsigned int)k2 * 2u + ((k2 >= 512) ? 16u : 0u);
  bf16x8 lo = *(const bf16x8*)(sB + o1);
  bf16x8 hi = *(const bf16x8*)(sB + o2);
  bf16x16 out;
#pragma unroll
  for (int i = 0; i < 8; ++i) { out[i] = lo[i]; out[i + 8] = hi[i]; }
  return out;
}

__device__ __forceinline__ floatx8 wmma_bf16(bf16x16 a, bf16x16 b, floatx8 c) {
  return __builtin_amdgcn_wmma_f32_16x16x32_bf16(false, a, false, b, (short)0, c,
                                                 false, false);
}

// ---------------------------------------------------------------------------
__global__ void __launch_bounds__(256)
f32_to_bf16(const float* __restrict__ in, __bf16* __restrict__ out, long n) {
  long i = (long)blockIdx.x * blockDim.x + threadIdx.x;
  long stride = (long)gridDim.x * blockDim.x;
  for (; i < n; i += stride) out[i] = (__bf16)in[i];
}

__global__ void __launch_bounds__(256)
init_state(float* __restrict__ c, float* __restrict__ h, __bf16* __restrict__ hb, int n) {
  int i = blockIdx.x * blockDim.x + threadIdx.x;
  if (i < n) { c[i] = 0.0f; h[i] = 0.0f; hb[i] = (__bf16)0.0f; }
}

// ---------------------------------------------------------------------------
// pre[s,b,g] = sum_i X[s,b,i] * Wx[g,i] + bx[g]
// M = 65536, N = 5120, K = 1024.  Block tile 128x64, wave tile 32x32.
// B (weights) staged by TDM into LDS once per block.
// ---------------------------------------------------------------------------
__global__ void __launch_bounds__(256)
gemm_pre_kernel(const __bf16* __restrict__ Xb, const __bf16* __restrict__ Wxb,
                const float* __restrict__ bx, float* __restrict__ pre) {
  __shared__ __align__(128) char sB[BT_BYTES];
  const int lane = threadIdx.x & 31;
  const int wave = threadIdx.x >> 5;
  const int wm = wave >> 1;
  const int wn = wave & 1;
  const int m0 = blockIdx.x * 128 + wm * 32;
  const int nBlk = blockIdx.y * 64;
  const int n0 = nBlk + wn * 32;

  if (wave == 0) {
    tdm_load_weight_tile_64x1024(Wxb + (size_t)nBlk * I_DIM,
                                 (unsigned int)(uintptr_t)sB);
    __builtin_amdgcn_s_wait_tensorcnt(0);
  }
  __syncthreads();

  floatx8 acc[2][2] = {};
#pragma unroll 1
  for (int k0 = 0; k0 < I_DIM; k0 += 32) {
    if (k0 + 64 < I_DIM) {
      __builtin_prefetch(Xb + (size_t)(m0 + (lane & 15)) * I_DIM + k0 + 64, 0, 1);
    }
    bf16x16 a0 = load_tile16x32(Xb, I_DIM, m0,      k0, lane);
    bf16x16 a1 = load_tile16x32(Xb, I_DIM, m0 + 16, k0, lane);
    bf16x16 b0 = lds_tile16x32(sB, wn * 32,      k0, lane);
    bf16x16 b1 = lds_tile16x32(sB, wn * 32 + 16, k0, lane);
    acc[0][0] = wmma_bf16(a0, b0, acc[0][0]);
    acc[0][1] = wmma_bf16(a0, b1, acc[0][1]);
    acc[1][0] = wmma_bf16(a1, b0, acc[1][0]);
    acc[1][1] = wmma_bf16(a1, b1, acc[1][1]);
  }

  const int mrow = (lane >> 4) << 3;
  const int ncol = lane & 15;
#pragma unroll
  for (int iS = 0; iS < 2; ++iS)
#pragma unroll
    for (int jS = 0; jS < 2; ++jS)
#pragma unroll
      for (int v = 0; v < 8; ++v) {
        int m = m0 + iS * 16 + mrow + v;
        int n = n0 + jS * 16 + ncol;
        pre[(size_t)m * G5 + n] = acc[iS][jS][v] + bx[n];
      }
}

// ---------------------------------------------------------------------------
// Step kernel A: G = sigmoid(pre[s][:, :4H] + h @ Wh^T + bh)
// M = 128, N = 4096, K = 1024.  64 blocks; Wh tile staged by TDM.
// ---------------------------------------------------------------------------
__global__ void __launch_bounds__(256)
lstm_gates_kernel(const __bf16* __restrict__ hb, const __bf16* __restrict__ Whb,
                  const float* __restrict__ bh, const float* __restrict__ pre_s,
                  const float* __restrict__ c, float* __restrict__ gates,
                  __bf16* __restrict__ a_bf) {
  __shared__ __align__(128) char sB[BT_BYTES];
  const int lane = threadIdx.x & 31;
  const int wave = threadIdx.x >> 5;
  const int wm = wave >> 1;
  const int wn = wave & 1;
  const int m0 = wm * 32;
  const int nBlk = blockIdx.x * 64;
  const int n0 = nBlk + wn * 32;

  if (wave == 0) {
    tdm_load_weight_tile_64x1024(Whb + (size_t)nBlk * H_DIM,
                                 (unsigned int)(uintptr_t)sB);
    __builtin_amdgcn_s_wait_tensorcnt(0);
  }
  __syncthreads();

  floatx8 acc[2][2] = {};
#pragma unroll 1
  for (int k0 = 0; k0 < H_DIM; k0 += 32) {
    bf16x16 a0 = load_tile16x32(hb, H_DIM, m0,      k0, lane);
    bf16x16 a1 = load_tile16x32(hb, H_DIM, m0 + 16, k0, lane);
    bf16x16 b0 = lds_tile16x32(sB, wn * 32,      k0, lane);
    bf16x16 b1 = lds_tile16x32(sB, wn * 32 + 16, k0, lane);
    acc[0][0] = wmma_bf16(a0, b0, acc[0][0]);
    acc[0][1] = wmma_bf16(a0, b1, acc[0][1]);
    acc[1][0] = wmma_bf16(a1, b0, acc[1][0]);
    acc[1][1] = wmma_bf16(a1, b1, acc[1][1]);
  }

  const int mrow = (lane >> 4) << 3;
  const int ncol = lane & 15;
#pragma unroll
  for (int iS = 0; iS < 2; ++iS)
#pragma unroll
    for (int jS = 0; jS < 2; ++jS)
#pragma unroll
      for (int v = 0; v < 8; ++v) {
        int b = m0 + iS * 16 + mrow + v;
        int n = n0 + jS * 16 + ncol;
        float g = sigmoidf_fast(acc[iS][jS][v] + pre_s[(size_t)b * G5 + n] + bh[n]);
        gates[b * G4 + n] = g;
        if (n >= 2 * H_DIM && n < 3 * H_DIM) {   // z gate -> peephole input
          int nz = n - 2 * H_DIM;
          a_bf[b * H_DIM + nz] = (__bf16)(g * tanhf(c[b * H_DIM + nz]));
        }
      }
}

// ---------------------------------------------------------------------------
// Step kernel B: u = tanh(pre[s][:, 4H:] + a @ Wum^T + bum)
//                c' = i*u + f*c ; h' = o*tanh(c')
// M = 128, N = 1024, K = 1024.  16 blocks; Wum tile staged by TDM.
// ---------------------------------------------------------------------------
__global__ void __launch_bounds__(256)
lstm_update_kernel(const __bf16* __restrict__ a_bf, const __bf16* __restrict__ Wumb,
                   const float* __restrict__ bum, const float* __restrict__ pre_s,
                   const float* __restrict__ gates, float* __restrict__ cbuf,
                   float* __restrict__ hbuf, __bf16* __restrict__ hb,
                   float* __restrict__ out_s) {
  __shared__ __align__(128) char sB[BT_BYTES];
  const int lane = threadIdx.x & 31;
  const int wave = threadIdx.x >> 5;
  const int wm = wave >> 1;
  const int wn = wave & 1;
  const int m0 = wm * 32;
  const int nBlk = blockIdx.x * 64;
  const int n0 = nBlk + wn * 32;

  if (wave == 0) {
    tdm_load_weight_tile_64x1024(Wumb + (size_t)nBlk * H_DIM,
                                 (unsigned int)(uintptr_t)sB);
    __builtin_amdgcn_s_wait_tensorcnt(0);
  }
  __syncthreads();

  floatx8 acc[2][2] = {};
#pragma unroll 1
  for (int k0 = 0; k0 < H_DIM; k0 += 32) {
    bf16x16 a0 = load_tile16x32(a_bf, H_DIM, m0,      k0, lane);
    bf16x16 a1 = load_tile16x32(a_bf, H_DIM, m0 + 16, k0, lane);
    bf16x16 b0 = lds_tile16x32(sB, wn * 32,      k0, lane);
    bf16x16 b1 = lds_tile16x32(sB, wn * 32 + 16, k0, lane);
    acc[0][0] = wmma_bf16(a0, b0, acc[0][0]);
    acc[0][1] = wmma_bf16(a0, b1, acc[0][1]);
    acc[1][0] = wmma_bf16(a1, b0, acc[1][0]);
    acc[1][1] = wmma_bf16(a1, b1, acc[1][1]);
  }

  const int mrow = (lane >> 4) << 3;
  const int ncol = lane & 15;
#pragma unroll
  for (int iS = 0; iS < 2; ++iS)
#pragma unroll
    for (int jS = 0; jS < 2; ++jS)
#pragma unroll
      for (int v = 0; v < 8; ++v) {
        int b = m0 + iS * 16 + mrow + v;
        int n = n0 + jS * 16 + ncol;
        float u = tanhf(acc[iS][jS][v] + pre_s[(size_t)b * G5 + G4 + n] + bum[n]);
        float gi = gates[b * G4 + n];
        float go = gates[b * G4 + H_DIM + n];
        float gf = gates[b * G4 + 3 * H_DIM + n];
        float cn = gi * u + gf * cbuf[b * H_DIM + n];
        float hn = go * tanhf(cn);
        cbuf[b * H_DIM + n] = cn;
        hbuf[b * H_DIM + n] = hn;
        hb[b * H_DIM + n] = (__bf16)hn;
        out_s[b * H_DIM + n] = hn;
      }
}

// ---------------------------------------------------------------------------
extern "C" void kernel_launch(void* const* d_in, const int* in_sizes, int n_in,
                              void* d_out, int out_size, void* d_ws, size_t ws_size,
                              hipStream_t stream) {
  const float* X   = (const float*)d_in[0];
  const float* Wx  = (const float*)d_in[1];
  const float* bx  = (const float*)d_in[2];
  const float* Wh  = (const float*)d_in[3];
  const float* bh  = (const float*)d_in[4];
  const float* Wum = (const float*)d_in[5];
  const float* bum = (const float*)d_in[6];

  float* out    = (float*)d_out;
  float* outSeq = out;
  float* hN     = out + (size_t)S_DIM * B_DIM * H_DIM;   // live h (== h_n)
  float* cN     = hN + (size_t)B_DIM * H_DIM;            // live c (== c_n)

  char* wp = (char*)d_ws;
  size_t off = 0;
  auto take = [&](size_t bytes) -> void* {
    void* p = wp + off;
    off += (bytes + 255) & ~(size_t)255;
    return p;
  };
  __bf16* Xb    = (__bf16*)take((size_t)S_DIM * B_DIM * I_DIM * 2);
  __bf16* Wxb   = (__bf16*)take((size_t)G5 * I_DIM * 2);
  __bf16* Whb   = (__bf16*)take((size_t)G4 * H_DIM * 2);
  __bf16* Wumb  = (__bf16*)take((size_t)H_DIM * H_DIM * 2);
  float*  pre   = (float*)take((size_t)S_DIM * B_DIM * G5 * 4);
  float*  gates = (float*)take((size_t)B_DIM * G4 * 4);
  __bf16* a_bf  = (__bf16*)take((size_t)B_DIM * H_DIM * 2);
  __bf16* hb    = (__bf16*)take((size_t)B_DIM * H_DIM * 2);

  f32_to_bf16<<<4096, 256, 0, stream>>>(X,   Xb,   (long)S_DIM * B_DIM * I_DIM);
  f32_to_bf16<<<2048, 256, 0, stream>>>(Wx,  Wxb,  (long)G5 * I_DIM);
  f32_to_bf16<<<2048, 256, 0, stream>>>(Wh,  Whb,  (long)G4 * H_DIM);
  f32_to_bf16<<<1024, 256, 0, stream>>>(Wum, Wumb, (long)H_DIM * H_DIM);
  init_state<<<(B_DIM * H_DIM + 255) / 256, 256, 0, stream>>>(cN, hN, hb,
                                                              B_DIM * H_DIM);

  gemm_pre_kernel<<<dim3((S_DIM * B_DIM) / 128, G5 / 64), 256, 0, stream>>>(
      Xb, Wxb, bx, pre);

  for (int s = 0; s < S_DIM; ++s) {
    const float* pre_s = pre + (size_t)s * B_DIM * G5;
    lstm_gates_kernel<<<G4 / 64, 256, 0, stream>>>(hb, Whb, bh, pre_s, cN, gates,
                                                   a_bf);
    lstm_update_kernel<<<H_DIM / 64, 256, 0, stream>>>(
        a_bf, Wumb, bum, pre_s, gates, cN, hN, hb,
        outSeq + (size_t)s * B_DIM * H_DIM);
  }
}
